// UnrolledAutoencodingTransformer_42760694399335
// MI455X (gfx1250) — matrix operand order repaired and checked by reading, not compiled, not run on hardware
//
#include <hip/hip_runtime.h>

typedef __attribute__((ext_vector_type(16))) _Float16 v16h;
typedef __attribute__((ext_vector_type(8)))  _Float16 v8h;
typedef __attribute__((ext_vector_type(8)))  float    v8f;
typedef __attribute__((ext_vector_type(4)))  int      v4i;

#define VOCAB 50257
#define VOCABP 50304   // padded to 393*128 (multiple of 64 too)
#define DIM 1024
#define TLEN 512
#define HID 4096
#define NB 4

// ---------- CDNA5 async global->LDS copy (guarded; fallback = reg pipeline) ----------
#ifndef USE_ASYNC_COPY
#if defined(__has_builtin)
#if __has_builtin(__builtin_amdgcn_global_load_async_to_lds_b128) && __has_builtin(__builtin_amdgcn_s_wait_asynccnt)
#define USE_ASYNC_COPY 1
#else
#define USE_ASYNC_COPY 0
#endif
#else
#define USE_ASYNC_COPY 0
#endif
#endif

#if USE_ASYNC_COPY
typedef __attribute__((address_space(1))) v4i gv4i;   // global 128-bit chunk
typedef __attribute__((address_space(3))) v4i lv4i;   // LDS 128-bit chunk
__device__ __forceinline__ void cp16_async(const _Float16* g, _Float16* l) {
    __builtin_amdgcn_global_load_async_to_lds_b128(
        (gv4i*)(unsigned long long)(uintptr_t)g,
        (lv4i*)(unsigned int)(uintptr_t)l,
        0, 0);
}
#endif

__device__ __forceinline__ float gelu_tanh(float x) {
    float x3 = x * x * x;
    return 0.5f * x * (1.0f + tanhf(0.7978845608028654f * (x + 0.044715f * x3)));
}

// ---------------- Encoder (only last token matters) ----------------
__global__ __launch_bounds__(256) void enc1_kernel(const int* __restrict__ ids,
                                                   const float* __restrict__ wte,
                                                   const float* __restrict__ w1,
                                                   const float* __restrict__ b1,
                                                   float* __restrict__ he) {
    int b = blockIdx.y;
    __shared__ float xs[DIM];
    int id = ids[b * TLEN + (TLEN - 1)];
    for (int k = threadIdx.x; k < DIM; k += 256) xs[k] = wte[(size_t)id * DIM + k];
    __syncthreads();
    int j = blockIdx.x * 256 + threadIdx.x;
    float acc = b1[j];
    for (int k = 0; k < DIM; ++k) acc += xs[k] * w1[(size_t)k * HID + j];
    he[b * HID + j] = gelu_tanh(acc);
}

__global__ __launch_bounds__(256) void enc2_kernel(const float* __restrict__ he,
                                                   const float* __restrict__ w2,
                                                   const float* __restrict__ b2,
                                                   float* __restrict__ e) {
    int b = blockIdx.y;
    __shared__ float hs[HID];
    for (int k = threadIdx.x; k < HID; k += 256) hs[k] = he[b * HID + k];
    __syncthreads();
    int i = blockIdx.x * 256 + threadIdx.x;
    float acc = b2[i];
    for (int j = 0; j < HID; ++j) acc += hs[j] * w2[(size_t)j * DIM + i];
    e[b * DIM + i] = acc;
}

// ---------------- windows gather -> f16 ----------------
__global__ __launch_bounds__(256) void window_kernel(const float* __restrict__ e,
                                                     _Float16* __restrict__ win) {
    int g = blockIdx.x * 256 + threadIdx.x;      // < 2048*512
    int m = g >> 9, k = g & 511;
    int b = m >> 9, t = m & 511;
    win[g] = (_Float16)e[(b << 10) + ((t + k) & (DIM - 1))];
}

// ---- transpose + convert: in f32 [K][N] -> out f16 [Np][K] (zero-fill n>=N) ----
__global__ __launch_bounds__(256) void transpose_cvt_kernel(const float* __restrict__ in,
                                                            _Float16* __restrict__ outT,
                                                            int K, int N) {
    __shared__ _Float16 s[64 * 72];
    int n0 = blockIdx.x * 64;
    int k0 = blockIdx.y * 64;
    int tid = threadIdx.x;
#pragma unroll
    for (int r = 0; r < 16; ++r) {
        int idx = tid + 256 * r;      // 0..4095
        int kr = idx >> 6;            // 0..63
        int nc = idx & 63;
        int n = n0 + nc;
        float v = (n < N) ? in[(size_t)(k0 + kr) * N + n] : 0.0f;
        s[nc * 72 + kr] = (_Float16)v;
    }
    __syncthreads();
#pragma unroll
    for (int r = 0; r < 2; ++r) {
        int idx = tid + 256 * r;      // 0..511
        int nr = idx >> 3;            // 0..63
        int seg = (idx & 7) * 8;
        *(v8h*)(outT + (size_t)(n0 + nr) * K + k0 + seg) = *(const v8h*)(s + nr * 72 + seg);
    }
}

// ---------------- Tiled WMMA GEMM ----------------
// C(M x N) = A(M x K) * B.  A: f16 row-major [m][K].
// B_F16T=true : B is f16 TRANSPOSED [n][K] (k contiguous) -> pure 16B staging.
// B_F16T=false: B is f32 [K][N_ld] staged with per-element guard (fallback only).
// Block tile 64x128, 8 waves (2x4) of 32x32, K-tile 32.
#define EP_BIAS_F16 1
#define EP_BIAS_GELU_F16 2
#define EP_TRANS_F32 3

template <bool B_F16T, int EP>
__global__ __launch_bounds__(256) void gemm_kernel(const _Float16* __restrict__ A,
                                                   const void* __restrict__ Bp,
                                                   const float* __restrict__ bias,
                                                   void* __restrict__ Cp,
                                                   int N_ld, int N_real, int K) {
    constexpr int LDT = 40;                      // padded f16 k-stride in LDS
    constexpr int BUF = (64 + 128) * LDT;        // halfs per stage buffer
#if USE_ASYNC_COPY
    constexpr int NSTAGE = B_F16T ? 2 : 1;
#else
    constexpr int NSTAGE = 1;
#endif
    constexpr size_t STAGE_BYTES = (size_t)NSTAGE * BUF * 2;
    constexpr size_t CT_BYTES = (EP == EP_TRANS_F32) ? (size_t)128 * 68 * 4 : 0;
    constexpr size_t SMEM_BYTES = STAGE_BYTES > CT_BYTES ? STAGE_BYTES : CT_BYTES;
    __shared__ __align__(16) char smem[SMEM_BYTES];
    _Float16* SH = (_Float16*)smem;

    const int tid = threadIdx.x;
    const int m0 = blockIdx.y * 64;
    const int bn0 = blockIdx.x * 128;
    const int w = tid >> 5, lane = tid & 31;
    const int wm = w & 1, wn = w >> 1;
    const int lr = lane & 15;
    const bool hi = (lane & 16) != 0;

    // staging thread mapping: 16B chunks, k-tile = 32 = 4 chunks/row
    const int a_m = tid >> 2;                    // 0..63
    const int kc = (tid & 3) * 8;                // 0,8,16,24
    const int a_lds = a_m * LDT + kc;
    const int b_lds0 = 64 * LDT + a_m * LDT + kc;       // B rows 0..63
    const int b_lds1 = b_lds0 + 64 * LDT;               // B rows 64..127
    const _Float16* Ag = A + (size_t)(m0 + a_m) * K + kc;
    const _Float16* Bg0 = nullptr;
    if (B_F16T) Bg0 = (const _Float16*)Bp + (size_t)(bn0 + a_m) * K + kc;

    v8f acc[2][2] = {};
    union Frag { v16h v; v8h h[2]; };

    auto compute = [&](const _Float16* base) {
        const _Float16* As = base;
        const _Float16* Bs = base + 64 * LDT;
        Frag af[2], bf[2];
        int klo = hi ? 8 : 0;   // A: interleaved K {klo..klo+7, klo+16..klo+23}
#pragma unroll
        for (int mi = 0; mi < 2; ++mi) {
            const _Float16* p = As + (wm * 32 + mi * 16 + lr) * LDT + klo;
            af[mi].h[0] = *(const v8h*)p;
            af[mi].h[1] = *(const v8h*)(p + 16);
        }
        int kb = hi ? 8 : 0;    // B: interleaved K {kb..kb+7, kb+16..kb+23}
#pragma unroll
        for (int ni = 0; ni < 2; ++ni) {
            const _Float16* p = Bs + (wn * 32 + ni * 16 + lr) * LDT + kb;
            bf[ni].h[0] = *(const v8h*)p;
            bf[ni].h[1] = *(const v8h*)(p + 16);
        }
#pragma unroll
        for (int mi = 0; mi < 2; ++mi)
#pragma unroll
            for (int ni = 0; ni < 2; ++ni)
                acc[mi][ni] = __builtin_amdgcn_wmma_f32_16x16x32_f16(
                    false, af[mi].v, false, bf[ni].v, (short)0, acc[mi][ni], false, false);
    };

    const int nt = K >> 5;
    if (B_F16T) {
#if USE_ASYNC_COPY
        // double-buffered async pipeline: 3 x b128 copies per thread per tile
        {
            cp16_async(Ag, SH + a_lds);
            cp16_async(Bg0, SH + b_lds0);
            cp16_async(Bg0 + (size_t)64 * K, SH + b_lds1);
        }
        for (int i = 0; i < nt; ++i) {
            int cur = i & 1;
            if (i + 1 < nt) {
                int kk = (i + 1) << 5;
                _Float16* nb = SH + (cur ^ 1) * BUF;
                cp16_async(Ag + kk, nb + a_lds);
                cp16_async(Bg0 + kk, nb + b_lds0);
                cp16_async(Bg0 + (size_t)64 * K + kk, nb + b_lds1);
                __builtin_amdgcn_s_wait_asynccnt(3);
            } else {
                __builtin_amdgcn_s_wait_asynccnt(0);
            }
            __syncthreads();
            compute(SH + cur * BUF);
            __syncthreads();
        }
#else
        // register-pipelined: prefetch next tile into VGPRs during compute
        v8h rA = *(const v8h*)Ag;
        v8h rB0 = *(const v8h*)Bg0;
        v8h rB1 = *(const v8h*)(Bg0 + (size_t)64 * K);
        for (int i = 0; i < nt; ++i) {
            *(v8h*)(SH + a_lds) = rA;
            *(v8h*)(SH + b_lds0) = rB0;
            *(v8h*)(SH + b_lds1) = rB1;
            __syncthreads();
            if (i + 1 < nt) {
                int kk = (i + 1) << 5;
                rA = *(const v8h*)(Ag + kk);
                rB0 = *(const v8h*)(Bg0 + kk);
                rB1 = *(const v8h*)(Bg0 + (size_t)64 * K + kk);
            }
            compute(SH);
            __syncthreads();
        }
#endif
    } else {
        // fallback: B f32 [K][N_ld], guarded scalar staging (cold path)
        const float* Bf = (const float*)Bp;
        for (int i = 0; i < nt; ++i) {
            int kk = i << 5;
            *(v8h*)(SH + a_lds) = *(const v8h*)(Ag + kk);
            _Float16* Bs = SH + 64 * LDT;
#pragma unroll
            for (int r = 0; r < 16; ++r) {
                int idx = tid + 256 * r;
                int k = idx >> 7;
                int n = idx & 127;
                int gn = bn0 + n;
                float v = (gn < N_real) ? Bf[(size_t)(kk + k) * N_ld + gn] : 0.0f;
                Bs[n * LDT + k] = (_Float16)v;
            }
            __syncthreads();
            compute(SH);
            __syncthreads();
        }
    }

    if (EP == EP_TRANS_F32) {
        // stage tile transposed in LDS, then coalesced (b, v, t) stores
        float* Ct = (float*)smem;   // [128 n][68]
#pragma unroll
        for (int mi = 0; mi < 2; ++mi)
#pragma unroll
            for (int ni = 0; ni < 2; ++ni)
#pragma unroll
                for (int r = 0; r < 8; ++r) {
                    int mm = wm * 32 + mi * 16 + r + (hi ? 8 : 0);
                    int nn = wn * 32 + ni * 16 + lr;
                    Ct[nn * 68 + mm] = acc[mi][ni][r];
                }
        __syncthreads();
        int bB = m0 >> 9;
        int t0 = m0 & 511;
        float* Of = (float*)Cp;
#pragma unroll
        for (int i = 0; i < 16; ++i) {
            int n = w + 8 * i;
            int v = bn0 + n;
            if (v < N_real) {
                size_t base = (((size_t)bB * N_real) + v) * TLEN + t0;
                Of[base + lane] = Ct[n * 68 + lane];
                Of[base + lane + 32] = Ct[n * 68 + lane + 32];
            }
        }
    } else {
        _Float16* Ch = (_Float16*)Cp;
#pragma unroll
        for (int mi = 0; mi < 2; ++mi)
#pragma unroll
            for (int ni = 0; ni < 2; ++ni)
#pragma unroll
                for (int r = 0; r < 8; ++r) {
                    int m = m0 + wm * 32 + mi * 16 + r + (hi ? 8 : 0);
                    int n = bn0 + wn * 32 + ni * 16 + lr;
                    float v = acc[mi][ni][r] + bias[n];
                    if (EP == EP_BIAS_GELU_F16) v = gelu_tanh(v);
                    Ch[(size_t)m * N_ld + n] = (_Float16)v;
                }
    }
}

// ---------------- log-softmax NLL ----------------
__global__ __launch_bounds__(256) void loss1_kernel(const float* __restrict__ logits,
                                                    const int* __restrict__ labels,
                                                    float* __restrict__ nll) {
    int b = blockIdx.x >> 5;           // 0..3
    int tc = blockIdx.x & 31;          // 32 chunks of 16 t
    int tl = threadIdx.x & 15;
    int g = threadIdx.x >> 4;          // 16 v-groups
    int t = tc * 16 + tl;
    const float* base = logits + ((size_t)b * VOCAB) * TLEN + t;
    float m = -1e30f, s = 0.0f;
    for (int v = g; v < VOCAB; v += 16) {
        float x = base[(size_t)v * TLEN];
        float nm = fmaxf(m, x);
        s = s * expf(m - nm) + expf(x - nm);
        m = nm;
    }
    __shared__ float sm[256], ss[256];
    sm[threadIdx.x] = m;
    ss[threadIdx.x] = s;
    __syncthreads();
    if (threadIdx.x < 16) {
        int tt = threadIdx.x;
        float M = -1e30f;
        for (int gg = 0; gg < 16; ++gg) M = fmaxf(M, sm[gg * 16 + tt]);
        float S = 0.0f;
        for (int gg = 0; gg < 16; ++gg) S += ss[gg * 16 + tt] * expf(sm[gg * 16 + tt] - M);
        int tg = tc * 16 + tt;
        int lab = labels[b * TLEN + tg];
        float ll = logits[(((size_t)b * VOCAB) + lab) * TLEN + tg];
        nll[b * TLEN + tg] = (M + logf(S)) - ll;
    }
}

__global__ __launch_bounds__(256) void loss2_kernel(const float* __restrict__ nll,
                                                    float* __restrict__ out0) {
    __shared__ float red[256];
    float a = 0.0f;
    for (int i = threadIdx.x; i < NB * TLEN; i += 256) a += nll[i];
    red[threadIdx.x] = a;
    __syncthreads();
    for (int off = 128; off; off >>= 1) {
        if (threadIdx.x < off) red[threadIdx.x] += red[threadIdx.x + off];
        __syncthreads();
    }
    if (threadIdx.x == 0) out0[0] = red[0] / (float)(NB * TLEN);
}

extern "C" void kernel_launch(void* const* d_in, const int* in_sizes, int n_in,
                              void* d_out, int out_size, void* d_ws, size_t ws_size,
                              hipStream_t stream) {
    (void)in_sizes; (void)n_in; (void)out_size;
    const int* ids      = (const int*)d_in[0];
    const int* labels   = (const int*)d_in[1];
    const float* wte    = (const float*)d_in[2];
    const float* enc_w1 = (const float*)d_in[3];
    const float* enc_b1 = (const float*)d_in[4];
    const float* enc_w2 = (const float*)d_in[5];
    const float* enc_b2 = (const float*)d_in[6];
    const float* proj_w = (const float*)d_in[7];
    const float* proj_b = (const float*)d_in[8];
    const float* dec_w1 = (const float*)d_in[9];
    const float* dec_b1 = (const float*)d_in[10];
    const float* dec_w2 = (const float*)d_in[11];
    const float* dec_b2 = (const float*)d_in[12];
    const float* lm_w   = (const float*)d_in[13];
    float* out = (float*)d_out;
    float* logits = out + 1;   // (B, V, T) f32

    char* ws = (char*)d_ws;
    size_t off = 0;
    auto alloc = [&](size_t bytes) { size_t p = off; off += (bytes + 255) & ~(size_t)255; return p; };
    float*    e_f32 = (float*)(ws + alloc((size_t)NB * DIM * 4));
    float*    he    = (float*)(ws + alloc((size_t)NB * HID * 4));
    float*    nll   = (float*)(ws + alloc((size_t)NB * TLEN * 4));
    _Float16* win_h = (_Float16*)(ws + alloc((size_t)2048 * 512 * 2));
    _Float16* p_h   = (_Float16*)(ws + alloc((size_t)2048 * 1024 * 2));
    _Float16* u_h   = (_Float16*)(ws + alloc((size_t)2048 * 4096 * 2));
    _Float16* d_h   = (_Float16*)(ws + alloc((size_t)2048 * 1024 * 2));
    _Float16* wproj = (_Float16*)(ws + alloc((size_t)1024 * 512 * 2));   // [n=1024][k=512]
    _Float16* wdec1 = (_Float16*)(ws + alloc((size_t)4096 * 1024 * 2));  // [n=4096][k=1024]
    _Float16* wdec2 = (_Float16*)(ws + alloc((size_t)1024 * 4096 * 2));  // [n=1024][k=4096]
    _Float16* lmw_t = (_Float16*)(ws + alloc((size_t)VOCABP * DIM * 2)); // [n=50304][k=1024]
    bool useF16LM = (off <= ws_size);

    // encoder on last token only
    enc1_kernel<<<dim3(HID / 256, NB), 256, 0, stream>>>(ids, wte, enc_w1, enc_b1, he);
    enc2_kernel<<<dim3(DIM / 256, NB), 256, 0, stream>>>(he, enc_w2, enc_b2, e_f32);
    window_kernel<<<(2048 * 512) / 256, 256, 0, stream>>>(e_f32, win_h);

    // one-time weight transpose+convert (B stored [n][K], k-contiguous)
    transpose_cvt_kernel<<<dim3(1024 / 64, 512 / 64), 256, 0, stream>>>(proj_w, wproj, 512, 1024);
    transpose_cvt_kernel<<<dim3(4096 / 64, 1024 / 64), 256, 0, stream>>>(dec_w1, wdec1, 1024, 4096);
    transpose_cvt_kernel<<<dim3(1024 / 64, 4096 / 64), 256, 0, stream>>>(dec_w2, wdec2, 4096, 1024);
    if (useF16LM)
        transpose_cvt_kernel<<<dim3(VOCABP / 64, 1024 / 64), 256, 0, stream>>>(lm_w, lmw_t, 1024, VOCAB);

    // p = windows @ proj_w + proj_b
    gemm_kernel<true, EP_BIAS_F16><<<dim3(1024 / 128, 2048 / 64), 256, 0, stream>>>(
        win_h, wproj, proj_b, p_h, 1024, 1024, 512);
    // u = gelu(p @ dec_w1 + dec_b1)
    gemm_kernel<true, EP_BIAS_GELU_F16><<<dim3(4096 / 128, 2048 / 64), 256, 0, stream>>>(
        p_h, wdec1, dec_b1, u_h, 4096, 4096, 1024);
    // d = u @ dec_w2 + dec_b2
    gemm_kernel<true, EP_BIAS_F16><<<dim3(1024 / 128, 2048 / 64), 256, 0, stream>>>(
        u_h, wdec2, dec_b2, d_h, 1024, 1024, 4096);
    // logits(b,v,t) = d @ lm_w, transposed store
    if (useF16LM) {
        gemm_kernel<true, EP_TRANS_F32><<<dim3(VOCABP / 128, 2048 / 64), 256, 0, stream>>>(
            d_h, lmw_t, nullptr, logits, VOCAB, VOCAB, 1024);
    } else {
        gemm_kernel<false, EP_TRANS_F32><<<dim3((VOCAB + 127) / 128, 2048 / 64), 256, 0, stream>>>(
            d_h, lm_w, nullptr, logits, VOCAB, VOCAB, 1024);
    }

    loss1_kernel<<<NB * 32, 256, 0, stream>>>(logits, labels, nll);
    loss2_kernel<<<1, 256, 0, stream>>>(nll, out);
}
